// SelfAttention_51324859187331
// MI455X (gfx1250) — matrix operand order repaired and checked by reading, not compiled, not run on hardware
//
#include <hip/hip_runtime.h>

#define NUM_HEADS 24
#define HD   128
#define SEQ  2048
#define CDIM 3072
#define QKVDIM 9216

typedef __attribute__((ext_vector_type(16))) __bf16 v16bf;
typedef __attribute__((ext_vector_type(8)))  __bf16 bf16x8;
typedef __attribute__((ext_vector_type(4)))  __bf16 bf16x4;
typedef __attribute__((ext_vector_type(8)))  float  v8f;
typedef __attribute__((ext_vector_type(4)))  float  f32x4;
typedef __attribute__((ext_vector_type(4)))  unsigned int u32x4;
typedef __attribute__((ext_vector_type(4)))  int    i32x4;
typedef __attribute__((ext_vector_type(8)))  int    i32x8;

__device__ __forceinline__ __bf16 f2bf(float f) {
  union { float f; unsigned u; } v; v.f = f;
  unsigned r = v.u + 0x7FFFu + ((v.u >> 16) & 1u);   // round-to-nearest-even
  unsigned short h = (unsigned short)(r >> 16);
  __bf16 o; __builtin_memcpy(&o, &h, 2); return o;
}

// ---------------------------------------------------------------------------
// Tensor Data Mover: 1-D contiguous copy of `nelem` 2-byte elements
// global -> LDS.  D# layout per CDNA5 ISA.
// ---------------------------------------------------------------------------
#if __has_builtin(__builtin_amdgcn_tensor_load_to_lds)
#define HAVE_TDM 1
__device__ __forceinline__ void tdm_load_1d(unsigned lds_addr, const void* gsrc,
                                            unsigned nelem /* <= 65535 */) {
  unsigned long long ga = (unsigned long long)(uintptr_t)gsrc;
  u32x4 g0;
  g0[0] = 1u;                                            // count=1 (valid user D#)
  g0[1] = lds_addr;                                      // lds_addr [63:32]
  g0[2] = (unsigned)ga;                                  // global_addr [95:64]
  g0[3] = (unsigned)((ga >> 32) & 0x01FFFFFFu)           // global_addr [120:96]
        | (2u << 30);                                    // type=2 ("image")
  i32x8 g1;
  g1[0] = (int)(1u << 16);                               // data_size=1 (2 bytes)
  g1[1] = (int)((nelem & 0xFFFFu) << 16);                // tensor_dim0[15:0]
  g1[2] = (int)((nelem >> 16) & 0xFFFFu) | (int)(1u << 16); // dim0 hi | tensor_dim1=1
  g1[3] = (int)((nelem & 0xFFFFu) << 16);                // tile_dim0 = nelem
  g1[4] = 1;                                             // tile_dim1 = 1
  g1[5] = (int)nelem;                                    // tensor_dim0_stride lo
  g1[6] = 0;
  g1[7] = 0;
  i32x4 z4 = {0, 0, 0, 0};
#if defined(__clang_major__) && (__clang_major__ >= 23)
  i32x8 z8 = {0, 0, 0, 0, 0, 0, 0, 0};
  __builtin_amdgcn_tensor_load_to_lds(g0, g1, z4, z4, z8, 0);
#else
  __builtin_amdgcn_tensor_load_to_lds(g0, g1, z4, z4, 0);
#endif
}
#else
#define HAVE_TDM 0
#endif

// ---------------------------------------------------------------------------
// Generic C[M,N] = A[M,K] * B[N,K]^T (+bias), A fp32 or bf16, B fp32 weights.
// Block tile 128x128, 256 threads = 8 waves (4 along M x 2 along N),
// per-wave 32x64 = 2x4 WMMA accumulators, K stepped by 32 (bf16 WMMA K).
// Double-buffered LDS + software-pipelined global loads: one barrier / step.
// ---------------------------------------------------------------------------
template<bool A_IS_BF16, bool ADD_BIAS>
__global__ __launch_bounds__(256)
void gemm_wmma(const void* __restrict__ Ap, const float* __restrict__ B,
               const float* __restrict__ bias, float* __restrict__ Cout,
               int M, int N, int K) {
  __shared__ __bf16 As[2][128][32];
  __shared__ __bf16 Bs[2][128][32];
  const int tid  = threadIdx.x;
  const int lane = tid & 31;
  const int wave = tid >> 5;
  const int wm   = wave & 3;      // 4 waves along M (32 rows each)
  const int wn   = wave >> 2;     // 2 waves along N (64 cols each)
  const int lh   = lane >> 4;     // lane half
  const int ll   = lane & 15;
  const int m0   = blockIdx.y * 128;
  const int n0   = blockIdx.x * 128;

  v8f acc[2][4];
#pragma unroll
  for (int mi = 0; mi < 2; ++mi)
#pragma unroll
    for (int nj = 0; nj < 4; ++nj)
      acc[mi][nj] = (v8f){0.f,0.f,0.f,0.f,0.f,0.f,0.f,0.f};

  f32x4  aF[4], bF[4];   // fp32 staging regs
  bf16x8 aB[2];          // bf16-A staging regs

  auto load_tiles = [&](int k0) {
    if (A_IS_BF16) {
      const __bf16* A = (const __bf16*)Ap;
#pragma unroll
      for (int c = 0; c < 2; ++c) {
        int idx = tid * 8 + c * 2048;
        int r = idx >> 5, cc = idx & 31;
        aB[c] = *(const bf16x8*)(A + (size_t)(m0 + r) * K + k0 + cc);
      }
    } else {
      const float* A = (const float*)Ap;
#pragma unroll
      for (int c = 0; c < 4; ++c) {
        int idx = tid * 4 + c * 1024;
        int r = idx >> 5, cc = idx & 31;
        aF[c] = *(const f32x4*)(A + (size_t)(m0 + r) * K + k0 + cc);
      }
    }
#pragma unroll
    for (int c = 0; c < 4; ++c) {
      int idx = tid * 4 + c * 1024;
      int r = idx >> 5, cc = idx & 31;
      bF[c] = *(const f32x4*)(B + (size_t)(n0 + r) * K + k0 + cc);
    }
  };
  auto store_tiles = [&](int buf) {
    if (A_IS_BF16) {
#pragma unroll
      for (int c = 0; c < 2; ++c) {
        int idx = tid * 8 + c * 2048;
        int r = idx >> 5, cc = idx & 31;
        *(bf16x8*)(&As[buf][r][cc]) = aB[c];
      }
    } else {
#pragma unroll
      for (int c = 0; c < 4; ++c) {
        int idx = tid * 4 + c * 1024;
        int r = idx >> 5, cc = idx & 31;
        bf16x4 b = { f2bf(aF[c][0]), f2bf(aF[c][1]), f2bf(aF[c][2]), f2bf(aF[c][3]) };
        *(bf16x4*)(&As[buf][r][cc]) = b;
      }
    }
#pragma unroll
    for (int c = 0; c < 4; ++c) {
      int idx = tid * 4 + c * 1024;
      int r = idx >> 5, cc = idx & 31;
      bf16x4 b = { f2bf(bF[c][0]), f2bf(bF[c][1]), f2bf(bF[c][2]), f2bf(bF[c][3]) };
      *(bf16x4*)(&Bs[buf][r][cc]) = b;
    }
  };

  load_tiles(0);
  store_tiles(0);
  __syncthreads();

  int cur = 0;
  for (int k0 = 0; k0 < K; k0 += 32) {
    const bool pf = (k0 + 32) < K;
    if (pf) load_tiles(k0 + 32);   // issue next-tile global loads early

    // A 16x32 layout: lane m = lane%16; K = lh*16..+15 contiguous
    v16bf af[2];
#pragma unroll
    for (int mi = 0; mi < 2; ++mi) {
      int row = wm * 32 + mi * 16 + ll;
      bf16x8 lo = *(const bf16x8*)(&As[cur][row][lh * 16]);
      bf16x8 hi = *(const bf16x8*)(&As[cur][row][lh * 16 + 8]);
      v16bf a;
#pragma unroll
      for (int e = 0; e < 8; ++e) { a[e] = lo[e]; a[e + 8] = hi[e]; }
      af[mi] = a;
    }
    // B 32x16 layout: lane n = lane%16; K = lh*16..+15 contiguous
    v16bf bfr[4];
#pragma unroll
    for (int nj = 0; nj < 4; ++nj) {
      int col = wn * 64 + nj * 16 + ll;
      bfr[nj] = *(const v16bf*)(&Bs[cur][col][lh * 16]);
    }
#pragma unroll
    for (int mi = 0; mi < 2; ++mi)
#pragma unroll
      for (int nj = 0; nj < 4; ++nj)
        acc[mi][nj] = __builtin_amdgcn_wmma_f32_16x16x32_bf16(
            false, af[mi], false, bfr[nj], (short)0, acc[mi][nj], false, false);

    if (pf) store_tiles(cur ^ 1);  // fill the other buffer
    __syncthreads();
    cur ^= 1;
  }

  // epilogue: C frag VGPR i -> row m = i + 8*lh, col n = lane%16
#pragma unroll
  for (int mi = 0; mi < 2; ++mi)
#pragma unroll
    for (int nj = 0; nj < 4; ++nj) {
      int col = n0 + wn * 64 + nj * 16 + ll;
      float bv = ADD_BIAS ? bias[col] : 0.f;
#pragma unroll
      for (int i = 0; i < 8; ++i) {
        int row = m0 + wm * 32 + mi * 16 + lh * 8 + i;
        Cout[(size_t)row * N + col] = acc[mi][nj][i] + bv;
      }
    }
}

// ---------------------------------------------------------------------------
// fp32 RMSNorm + RoPE for q,k; v passthrough. One block per (head, token).
// Writes bf16 q/k/v in [H][L][D]; q pre-scaled by 1/sqrt(D).
// ---------------------------------------------------------------------------
__global__ __launch_bounds__(128)
void norm_rope(const float* __restrict__ qkv, const float* __restrict__ pe,
               const float* __restrict__ q_scale, const float* __restrict__ k_scale,
               __bf16* __restrict__ qb, __bf16* __restrict__ kb,
               __bf16* __restrict__ vb) {
  const int h = blockIdx.x;
  const int l = blockIdx.y;
  const int d = threadIdx.x;
  const size_t base = (size_t)l * QKVDIM + h * HD + d;
  float qv = qkv[base];
  float kv = qkv[base + CDIM];
  float vv = qkv[base + 2 * CDIM];

  float s0 = qv * qv, s1 = kv * kv;
#pragma unroll
  for (int off = 1; off < 32; off <<= 1) {
    s0 += __shfl_xor(s0, off);
    s1 += __shfl_xor(s1, off);
  }
  __shared__ float redq[4], redk[4];
  const int w = threadIdx.x >> 5, lane = threadIdx.x & 31;
  if (lane == 0) { redq[w] = s0; redk[w] = s1; }
  __syncthreads();
  float sq = redq[0] + redq[1] + redq[2] + redq[3];
  float sk = redk[0] + redk[1] + redk[2] + redk[3];
  float rq = rsqrtf(sq * (1.f / HD) + 1e-6f);
  float rk = rsqrtf(sk * (1.f / HD) + 1e-6f);
  float qn = qv * rq * q_scale[d];
  float kn = kv * rk * k_scale[d];

  __shared__ float sQ[HD], sK[HD];
  sQ[d] = qn; sK[d] = kn;
  __syncthreads();
  const int i = d >> 1, r = d & 1;
  const size_t pb = (((size_t)l * (HD / 2) + i) * 2 + r) * 2;  // [L][D/2][2][2]
  float p0 = pe[pb], p1 = pe[pb + 1];
  float qr = p0 * sQ[2 * i] + p1 * sQ[2 * i + 1];
  float kr = p0 * sK[2 * i] + p1 * sK[2 * i + 1];

  const size_t ob = ((size_t)h * SEQ + l) * HD + d;
  qb[ob] = f2bf(qr * 0.08838834764831845f);  // fold in 1/sqrt(128)
  kb[ob] = f2bf(kr);
  vb[ob] = f2bf(vv);
}

// ---------------------------------------------------------------------------
// Flash attention: block = (head, 128 q rows), 8 waves x 16 rows each.
// K tiles: TDM (tensor_load_to_lds) double-buffered, synced with
// s_wait_tensorcnt + barrier.  V tiles: software-pipelined global loads,
// transposed into LDS.  Online softmax; WMMA for QK^T and PV.
// ---------------------------------------------------------------------------
__global__ __launch_bounds__(256)
void flash_attn(const __bf16* __restrict__ qb, const __bf16* __restrict__ kbm,
                const __bf16* __restrict__ vbm, __bf16* __restrict__ aout) {
  __shared__ __bf16 Ks[2][64][128];   // [buf][key][d]
  __shared__ __bf16 Vt[2][128][64];   // [buf][d][key]
  __shared__ __bf16 Pb[8][16][64];    // per-wave P scratch [m][key]

  const int h = blockIdx.x, qt = blockIdx.y;
  const int tid = threadIdx.x, lane = tid & 31, wave = tid >> 5;
  const int lh = lane >> 4, ll = lane & 15;
  const __bf16* Qh = qb  + (size_t)h * SEQ * HD;
  const __bf16* Kh = kbm + (size_t)h * SEQ * HD;
  const __bf16* Vh = vbm + (size_t)h * SEQ * HD;
  const int q0 = qt * 128 + wave * 16;
  const int NT = SEQ / 64;

  // Q A-fragments (held in registers for the whole kernel)
  v16bf qf[4];
  const __bf16* qrow = Qh + (size_t)(q0 + ll) * HD;
#pragma unroll
  for (int dc = 0; dc < 4; ++dc) {
    bf16x8 lo = *(const bf16x8*)(qrow + dc * 32 + lh * 16);
    bf16x8 hi = *(const bf16x8*)(qrow + dc * 32 + lh * 16 + 8);
    v16bf a;
#pragma unroll
    for (int e = 0; e < 8; ++e) { a[e] = lo[e]; a[e + 8] = hi[e]; }
    qf[dc] = a;
  }

  v8f O[8];
#pragma unroll
  for (int t = 0; t < 8; ++t) O[t] = (v8f){0.f,0.f,0.f,0.f,0.f,0.f,0.f,0.f};
  float rmax[8], rsum[8];
#pragma unroll
  for (int i = 0; i < 8; ++i) { rmax[i] = -1e30f; rsum[i] = 0.f; }

  bf16x8 vr[4];  // V staging registers

  auto k_stage = [&](int kt, int buf) {   // K tile: contiguous 8192 bf16
#if HAVE_TDM
    if (wave == 0)
      tdm_load_1d((unsigned)(uintptr_t)&Ks[buf][0][0],
                  Kh + (size_t)kt * 64 * HD, 64 * HD);
#else
#pragma unroll
    for (int c = 0; c < 4; ++c) {
      int idx = tid * 8 + c * 2048;
      int r = idx >> 7, cc = idx & 127;
      *(bf16x8*)(&Ks[buf][r][cc]) =
          *(const bf16x8*)(Kh + (size_t)(kt * 64 + r) * HD + cc);
    }
#endif
  };
  auto k_wait = [&]() {
#if HAVE_TDM
    if (wave == 0) __builtin_amdgcn_s_wait_tensorcnt(0);
#endif
  };
  auto v_load = [&](int kt) {
#pragma unroll
    for (int c = 0; c < 4; ++c) {
      int idx = tid * 8 + c * 2048;
      int r = idx >> 7, cc = idx & 127;
      vr[c] = *(const bf16x8*)(Vh + (size_t)(kt * 64 + r) * HD + cc);
    }
  };
  auto v_scatter = [&](int buf) {
#pragma unroll
    for (int c = 0; c < 4; ++c) {
      int idx = tid * 8 + c * 2048;
      int r = idx >> 7, cc = idx & 127;
#pragma unroll
      for (int e = 0; e < 8; ++e) Vt[buf][cc + e][r] = vr[c][e];
    }
  };

  // prologue: stage tile 0
  k_stage(0, 0);
  v_load(0);
  v_scatter(0);
  k_wait();
  __syncthreads();

  int cur = 0;
  for (int kt = 0; kt < NT; ++kt) {
    const bool pf = (kt + 1) < NT;
    if (pf) {
      k_stage(kt + 1, cur ^ 1);  // TDM into other buffer, runs during compute
      v_load(kt + 1);            // global V loads in flight during compute
    }

    // S = Q K^T : 4 N-tiles of 16 keys, K-dim = 128 in 4 chunks of 32
    v8f S[4];
#pragma unroll
    for (int j = 0; j < 4; ++j) {
      v8f s = (v8f){0.f,0.f,0.f,0.f,0.f,0.f,0.f,0.f};
#pragma unroll
      for (int dc = 0; dc < 4; ++dc) {
        v16bf b = *(const v16bf*)(&Ks[cur][j * 16 + ll][dc * 32 + lh * 16]);
        s = __builtin_amdgcn_wmma_f32_16x16x32_bf16(
            false, qf[dc], false, b, (short)0, s, false, false);
      }
      S[j] = s;
    }

    // online softmax over this 64-key tile (rows live in 16-lane halves)
    float pcur[4][8];
#pragma unroll
    for (int i = 0; i < 8; ++i) {
      float tmax = fmaxf(fmaxf(S[0][i], S[1][i]), fmaxf(S[2][i], S[3][i]));
#pragma unroll
      for (int off = 1; off < 16; off <<= 1)
        tmax = fmaxf(tmax, __shfl_xor(tmax, off));
      float mnew = fmaxf(rmax[i], tmax);
      float corr = __expf(rmax[i] - mnew);
      rmax[i] = mnew;
      float rs = 0.f;
#pragma unroll
      for (int j = 0; j < 4; ++j) {
        float p = __expf(S[j][i] - mnew);
        pcur[j][i] = p;
        rs += p;
      }
#pragma unroll
      for (int off = 1; off < 16; off <<= 1) rs += __shfl_xor(rs, off);
      rsum[i] = rsum[i] * corr + rs;
#pragma unroll
      for (int t = 0; t < 8; ++t) O[t][i] *= corr;
    }

    // C-layout P -> LDS -> A-layout (per-wave scratch; LDS in-order per wave)
#pragma unroll
    for (int j = 0; j < 4; ++j)
#pragma unroll
      for (int i = 0; i < 8; ++i)
        Pb[wave][i + 8 * lh][j * 16 + ll] = f2bf(pcur[j][i]);

    // O += P V : key-dim 64 in 2 chunks of 32, 8 d-tiles of 16
#pragma unroll
    for (int kc = 0; kc < 2; ++kc) {
      bf16x8 lo = *(const bf16x8*)(&Pb[wave][ll][kc * 32 + lh * 16]);
      bf16x8 hi = *(const bf16x8*)(&Pb[wave][ll][kc * 32 + lh * 16 + 8]);
      v16bf pa;
#pragma unroll
      for (int e = 0; e < 8; ++e) { pa[e] = lo[e]; pa[e + 8] = hi[e]; }
#pragma unroll
      for (int t = 0; t < 8; ++t) {
        v16bf vf = *(const v16bf*)(&Vt[cur][t * 16 + ll][kc * 32 + lh * 16]);
        O[t] = __builtin_amdgcn_wmma_f32_16x16x32_bf16(
            false, pa, false, vf, (short)0, O[t], false, false);
      }
    }

    if (pf) v_scatter(cur ^ 1);  // publish next V tile into the other buffer
    k_wait();                    // wave0: TDM for next K tile complete
    __syncthreads();
    cur ^= 1;
  }

  // normalize and emit bf16 [L][C] (C index = h*128 + d)
#pragma unroll
  for (int t = 0; t < 8; ++t)
#pragma unroll
    for (int i = 0; i < 8; ++i) {
      int row = q0 + 8 * lh + i;
      int col = h * HD + t * 16 + ll;
      aout[(size_t)row * CDIM + col] = f2bf(O[t][i] / rsum[i]);
    }
}

// ---------------------------------------------------------------------------
extern "C" void kernel_launch(void* const* d_in, const int* in_sizes, int n_in,
                              void* d_out, int out_size, void* d_ws, size_t ws_size,
                              hipStream_t stream) {
  const float* x       = (const float*)d_in[0];
  const float* pe      = (const float*)d_in[1];
  const float* w_qkv   = (const float*)d_in[2];
  const float* q_scale = (const float*)d_in[3];
  const float* k_scale = (const float*)d_in[4];
  const float* w_proj  = (const float*)d_in[5];
  const float* b_proj  = (const float*)d_in[6];
  float* out = (float*)d_out;

  char* ws = (char*)d_ws;
  float*  qkv = (float*)ws;                                   // 2048*9216 f32
  size_t  off = (size_t)SEQ * QKVDIM * sizeof(float);
  __bf16* qb  = (__bf16*)(ws + off); off += (size_t)NUM_HEADS * SEQ * HD * 2;
  __bf16* kb  = (__bf16*)(ws + off); off += (size_t)NUM_HEADS * SEQ * HD * 2;
  __bf16* vb  = (__bf16*)(ws + off); off += (size_t)NUM_HEADS * SEQ * HD * 2;
  __bf16* abf = (__bf16*)ws;  // aliases qkv region (dead after norm_rope)

  // 1) qkv = x @ w_qkv^T  (fp32 out for accurate RMSNorm)
  gemm_wmma<false, false><<<dim3(QKVDIM / 128, SEQ / 128), 256, 0, stream>>>(
      (const void*)x, w_qkv, nullptr, qkv, SEQ, QKVDIM, CDIM);
  // 2) RMSNorm + RoPE, split into bf16 q/k/v [H][L][D]
  norm_rope<<<dim3(NUM_HEADS, SEQ), 128, 0, stream>>>(
      qkv, pe, q_scale, k_scale, qb, kb, vb);
  // 3) flash attention -> bf16 [L][C]
  flash_attn<<<dim3(NUM_HEADS, SEQ / 128), 256, 0, stream>>>(qb, kb, vb, abf);
  // 4) out = attn @ w_proj^T + b_proj (fp32)
  gemm_wmma<true, true><<<dim3(CDIM / 128, SEQ / 128), 256, 0, stream>>>(
      (const void*)abf, w_proj, b_proj, out, SEQ, CDIM, CDIM);
}
